// E_GCL_21560735826057
// MI455X (gfx1250) — matrix operand order repaired and checked by reading, not compiled, not run on hardware
//
#include <hip/hip_runtime.h>
#include <hip/hip_bf16.h>

typedef __attribute__((ext_vector_type(16))) _Float16 v16h;
typedef __attribute__((ext_vector_type(8)))  _Float16 v8h;
typedef __attribute__((ext_vector_type(8)))  float    v8f;

// ---------------------------------------------------------------------------
// Utility kernels
// ---------------------------------------------------------------------------
__global__ void zero_f32(float* p, size_t n) {
    size_t i = (size_t)blockIdx.x * blockDim.x + threadIdx.x;
    if (i < n) p[i] = 0.0f;
}

// Convert f32 weight [src_rows,128] -> f16 [dst_rows,128], zero-padding extra rows.
__global__ void cvt_pad(const float* __restrict__ src, _Float16* __restrict__ dst,
                        int src_rows, int dst_rows) {
    int i = blockIdx.x * blockDim.x + threadIdx.x;
    int total = dst_rows * 128;
    if (i >= total) return;
    int r = i >> 7;
    dst[i] = (r < src_rows) ? (_Float16)src[i] : (_Float16)0.0f;
}

// ---------------------------------------------------------------------------
// Multi-tile WMMA helper (wave32, v_wmma_f32_16x16x32_f16).
// Each wave computes MT stacked 16x16 output tiles sharing one 16-wide N slab,
// so each B fragment (global, L2-resident weights) is reused MT times.
//
// A (MT*16 x K f16) in LDS row-major, row stride lda.
//   ISA A layout: lane = M (lane&15); VGPR0..3 = K{0..7} (lanes 0-15) or
//   K{8..15} (lanes 16-31); VGPR4..7 same +16 -> two 16B LDS loads per tile.
// B (K x 16 f16) in global row-major [K,128].
//   ISA B layout: lane = N (lane&15); halves 0..7 = K{0..7} (lanes 0-15) or
//   K{8..15} (lanes 16-31); halves 8..15 same +16 -> strided scalar loads.
// ---------------------------------------------------------------------------
template <int MT>
__device__ __forceinline__ void wmma_tiles(const _Float16* A, int lda,
                                           const _Float16* __restrict__ B,
                                           int n0, int K, int lane, v8f* acc) {
    const int m  = lane & 15;
    const int kb = (lane < 16) ? 0 : 8;
    for (int k0 = 0; k0 < K; k0 += 32) {
        v16h b;
#pragma unroll
        for (int i = 0; i < 8; ++i) {
            b[i]     = B[(size_t)(k0 + kb + i) * 128 + n0 + m];
            b[i + 8] = B[(size_t)(k0 + kb + 16 + i) * 128 + n0 + m];
        }
#pragma unroll
        for (int t = 0; t < MT; ++t) {
            const _Float16* pa = A + (t * 16 + m) * lda + k0 + kb;
            v8h lo = *(const v8h*)(pa);
            v8h hi = *(const v8h*)(pa + 16);
            v16h a;
#pragma unroll
            for (int i = 0; i < 8; ++i) { a[i] = lo[i]; a[i + 8] = hi[i]; }
            acc[t] = __builtin_amdgcn_wmma_f32_16x16x32_f16(
                false, a, false, b, (short)0, acc[t], false, false);
        }
    }
}

__device__ __forceinline__ float silu(float x) {
    return x / (1.0f + __expf(-x));
}

// ---------------------------------------------------------------------------
// Edge kernel: 64 edges per block, 256 threads (8 waves).
// Wave w owns output features [16w, 16w+16) for all 4 M-subtiles.
// ---------------------------------------------------------------------------
__global__ void __launch_bounds__(256)
egcl_edge_kernel(const float* __restrict__ h, const float* __restrict__ coord,
                 const int* __restrict__ ei,
                 const float* __restrict__ b_e1, const float* __restrict__ b_e2,
                 const float* __restrict__ b_c1, const float* __restrict__ w_c2,
                 const _Float16* __restrict__ we1h, const _Float16* __restrict__ we2h,
                 const _Float16* __restrict__ wc1h,
                 float* __restrict__ ef_out, float* __restrict__ agg,
                 float* __restrict__ tsum, float* __restrict__ cnt, int E) {
    __shared__ __align__(16) _Float16 As[64 * 288];  // [h_row | h_col | radial | pad]
    __shared__ __align__(16) _Float16 Hs[64 * 128];  // hidden1, then edge_feat (f16)
    __shared__ __align__(16) float    Cs[64 * 128];  // coord-MLP hidden (f32)
    __shared__ float cds[64 * 3];
    __shared__ int   rows[64];
    __shared__ int   cols[64];

    const int e0   = blockIdx.x * 64;
    const int tid  = threadIdx.x;
    const int lane = tid & 31;
    const int wv   = tid >> 5;
    const int n0   = wv * 16;

    if (tid < 64) {
        int e = e0 + tid;
        int r = ei[e];
        int c = ei[E + e];
        rows[tid] = r;
        cols[tid] = c;
        float dx = coord[3 * r + 0] - coord[3 * c + 0];
        float dy = coord[3 * r + 1] - coord[3 * c + 1];
        float dz = coord[3 * r + 2] - coord[3 * c + 2];
        cds[tid * 3 + 0] = dx;
        cds[tid * 3 + 1] = dy;
        cds[tid * 3 + 2] = dz;
        As[tid * 288 + 256] = (_Float16)(dx * dx + dy * dy + dz * dz);
#pragma unroll
        for (int k = 257; k < 288; ++k) As[tid * 288 + k] = (_Float16)0.0f;
    }
    __syncthreads();

    // Gather h[row] || h[col] into the f16 A tile.
    for (int i = tid; i < 64 * 256; i += 256) {
        int m = i >> 8;
        int k = i & 255;
        int node = (k < 128) ? rows[m] : cols[m];
        As[m * 288 + k] = (_Float16)h[(size_t)node * 128 + (k & 127)];
    }
    __syncthreads();

    // Edge MLP layer 1: [64,288] x [288,16] + bias, silu
    v8f acc1[4] = {};
    wmma_tiles<4>(As, 288, we1h, n0, 288, lane, acc1);
#pragma unroll
    for (int t = 0; t < 4; ++t)
#pragma unroll
        for (int r = 0; r < 8; ++r) {
            int mm = t * 16 + r + ((lane >> 4) << 3);
            int nn = n0 + (lane & 15);
            Hs[mm * 128 + nn] = (_Float16)silu(acc1[t][r] + b_e1[nn]);
        }
    __syncthreads();

    // Edge MLP layer 2: + bias, silu -> edge_feat
    v8f acc2[4] = {};
    wmma_tiles<4>(Hs, 128, we2h, n0, 128, lane, acc2);
    __syncthreads();  // all waves done reading Hs before overwrite
#pragma unroll
    for (int t = 0; t < 4; ++t)
#pragma unroll
        for (int r = 0; r < 8; ++r) {
            int mm = t * 16 + r + ((lane >> 4) << 3);
            int nn = n0 + (lane & 15);
            float x = silu(acc2[t][r] + b_e2[nn]);
            Hs[mm * 128 + nn] = (_Float16)x;                  // for coord MLP
            ef_out[(size_t)(e0 + mm) * 128 + nn] = x;         // output #3
            atomicAdd(&agg[(size_t)rows[mm] * 128 + nn], x);  // segment_sum
        }
    __syncthreads();

    // Coord MLP layer 1: silu(edge_feat @ w_c1 + b_c1)
    v8f acc3[4] = {};
    wmma_tiles<4>(Hs, 128, wc1h, n0, 128, lane, acc3);
#pragma unroll
    for (int t = 0; t < 4; ++t)
#pragma unroll
        for (int r = 0; r < 8; ++r) {
            int mm = t * 16 + r + ((lane >> 4) << 3);
            int nn = n0 + (lane & 15);
            Cs[mm * 128 + nn] = silu(acc3[t][r] + b_c1[nn]);
        }
    __syncthreads();

    // Coord MLP layer 2 (128 -> 1) + trans scatter
    if (tid < 64) {
        float s = 0.0f;
        for (int k = 0; k < 128; ++k) s += Cs[tid * 128 + k] * w_c2[k];
        int rn = rows[tid];
        atomicAdd(&tsum[(size_t)rn * 3 + 0], cds[tid * 3 + 0] * s);
        atomicAdd(&tsum[(size_t)rn * 3 + 1], cds[tid * 3 + 1] * s);
        atomicAdd(&tsum[(size_t)rn * 3 + 2], cds[tid * 3 + 2] * s);
        atomicAdd(&cnt[rn], 1.0f);
    }
}

// ---------------------------------------------------------------------------
// Node kernel: 64 nodes per block, 256 threads (8 waves). N may not divide 64;
// guards wrap only scalar loads/stores (WMMA stays uniform, EXEC all-ones).
// ---------------------------------------------------------------------------
__global__ void __launch_bounds__(256)
egcl_node_kernel(const float* __restrict__ h, const float* __restrict__ coord,
                 const float* __restrict__ agg, const float* __restrict__ tsum,
                 const float* __restrict__ cnt,
                 const float* __restrict__ b_n1, const float* __restrict__ b_n2,
                 const _Float16* __restrict__ wn1h, const _Float16* __restrict__ wn2h,
                 float* __restrict__ hnew, float* __restrict__ cnew, int N) {
    __shared__ __align__(16) _Float16 As[64 * 256];  // [h | agg] f16
    __shared__ __align__(16) _Float16 Hs[64 * 128];

    const int m0   = blockIdx.x * 64;
    const int tid  = threadIdx.x;
    const int lane = tid & 31;
    const int wv   = tid >> 5;
    const int n0   = wv * 16;

    for (int i = tid; i < 64 * 256; i += 256) {
        int m = i >> 8;
        int k = i & 255;
        int node = m0 + m;
        if (node >= N) node = N - 1;  // clamp; padded rows never stored
        float v = (k < 128) ? h[(size_t)node * 128 + k]
                            : agg[(size_t)node * 128 + (k - 128)];
        As[m * 256 + k] = (_Float16)v;
    }
    __syncthreads();

    // Node MLP layer 1: [64,256] x [256,16] + bias, silu
    v8f acc1[4] = {};
    wmma_tiles<4>(As, 256, wn1h, n0, 256, lane, acc1);
#pragma unroll
    for (int t = 0; t < 4; ++t)
#pragma unroll
        for (int r = 0; r < 8; ++r) {
            int mm = t * 16 + r + ((lane >> 4) << 3);
            int nn = n0 + (lane & 15);
            Hs[mm * 128 + nn] = (_Float16)silu(acc1[t][r] + b_n1[nn]);
        }
    __syncthreads();

    // Node MLP layer 2 + residual
    v8f acc2[4] = {};
    wmma_tiles<4>(Hs, 128, wn2h, n0, 128, lane, acc2);
#pragma unroll
    for (int t = 0; t < 4; ++t)
#pragma unroll
        for (int r = 0; r < 8; ++r) {
            int mm = t * 16 + r + ((lane >> 4) << 3);
            int nn = n0 + (lane & 15);
            if (m0 + mm < N) {
                size_t idx = (size_t)(m0 + mm) * 128 + nn;
                hnew[idx] = h[idx] + acc2[t][r] + b_n2[nn];
            }
        }

    // coord_new = coord + t_sum / max(cnt, 1)
    if (tid < 64 && m0 + tid < N) {
        int node = m0 + tid;
        float c = fmaxf(cnt[node], 1.0f);
#pragma unroll
        for (int j = 0; j < 3; ++j)
            cnew[(size_t)node * 3 + j] =
                coord[(size_t)node * 3 + j] + tsum[(size_t)node * 3 + j] / c;
    }
}

// ---------------------------------------------------------------------------
// Launch
// ---------------------------------------------------------------------------
extern "C" void kernel_launch(void* const* d_in, const int* in_sizes, int n_in,
                              void* d_out, int out_size, void* d_ws, size_t ws_size,
                              hipStream_t stream) {
    const float* h     = (const float*)d_in[0];
    const float* coord = (const float*)d_in[1];
    const int*   ei    = (const int*)d_in[2];
    const float* w_e1  = (const float*)d_in[3];
    const float* b_e1  = (const float*)d_in[4];
    const float* w_e2  = (const float*)d_in[5];
    const float* b_e2  = (const float*)d_in[6];
    const float* w_n1  = (const float*)d_in[7];
    const float* b_n1  = (const float*)d_in[8];
    const float* w_n2  = (const float*)d_in[9];
    const float* b_n2  = (const float*)d_in[10];
    const float* w_c1  = (const float*)d_in[11];
    const float* b_c1  = (const float*)d_in[12];
    const float* w_c2  = (const float*)d_in[13];

    const int N = in_sizes[0] / 128;
    const int E = in_sizes[2] / 2;

    // Workspace layout
    char* ws = (char*)d_ws;
    float* agg  = (float*)ws;  ws += (size_t)N * 128 * sizeof(float);
    float* tsum = (float*)ws;  ws += (size_t)N * 3   * sizeof(float);
    float* cnt  = (float*)ws;  ws += (size_t)N       * sizeof(float);
    _Float16* we1h = (_Float16*)ws;  ws += (size_t)288 * 128 * 2;
    _Float16* we2h = (_Float16*)ws;  ws += (size_t)128 * 128 * 2;
    _Float16* wc1h = (_Float16*)ws;  ws += (size_t)128 * 128 * 2;
    _Float16* wn1h = (_Float16*)ws;  ws += (size_t)256 * 128 * 2;
    _Float16* wn2h = (_Float16*)ws;  ws += (size_t)128 * 128 * 2;

    // Zero accumulators (agg | tsum | cnt contiguous: N*132 floats)
    {
        size_t nz = (size_t)N * 132;
        zero_f32<<<(int)((nz + 255) / 256), 256, 0, stream>>>(agg, nz);
    }
    // Weight conversions (f32 -> f16, K-padded row-major [K,128])
    cvt_pad<<<(288 * 128 + 255) / 256, 256, 0, stream>>>(w_e1, we1h, 257, 288);
    cvt_pad<<<(128 * 128 + 255) / 256, 256, 0, stream>>>(w_e2, we2h, 128, 128);
    cvt_pad<<<(128 * 128 + 255) / 256, 256, 0, stream>>>(w_c1, wc1h, 128, 128);
    cvt_pad<<<(256 * 128 + 255) / 256, 256, 0, stream>>>(w_n1, wn1h, 256, 256);
    cvt_pad<<<(128 * 128 + 255) / 256, 256, 0, stream>>>(w_n2, wn2h, 128, 128);

    // Output layout: h_new [N,128] | coord_new [N,3] | edge_feat [E,128]
    float* hnew = (float*)d_out;
    float* cnew = hnew + (size_t)N * 128;
    float* ef   = cnew + (size_t)N * 3;

    egcl_edge_kernel<<<E / 64, 256, 0, stream>>>(h, coord, ei, b_e1, b_e2, b_c1,
                                                 w_c2, we1h, we2h, wc1h,
                                                 ef, agg, tsum, cnt, E);
    egcl_node_kernel<<<(N + 63) / 64, 256, 0, stream>>>(h, coord, agg, tsum, cnt,
                                                        b_n1, b_n2, wn1h, wn2h,
                                                        hnew, cnew, N);
}
